// MultiHeadAttention_44109314130326
// MI455X (gfx1250) — matrix operand order repaired and checked by reading, not compiled, not run on hardware
//
#include <hip/hip_runtime.h>

// Problem constants (from the reference)
constexpr int Bn = 4;      // batch
constexpr int Cn = 1024;   // channels
constexpr int Tn = 2048;   // sequence
constexpr int Hn = 16;     // heads
constexpr int Dn = 64;     // head dim
// softmax scale = 1/sqrt(D) = 1/8

typedef _Float16 f16;
typedef __attribute__((ext_vector_type(16))) _Float16 v16h;
typedef __attribute__((ext_vector_type(8)))  _Float16 v8h;
typedef __attribute__((ext_vector_type(8)))  float    v8f;
typedef __attribute__((ext_vector_type(4)))  int      v4i;

// ---------------------------------------------------------------------------
// Async global -> LDS path (CDNA5 GLOBAL_LOAD_ASYNC_TO_LDS_B128 + ASYNCcnt),
// guarded so the file compiles on toolchains without the builtins.
// Builtin signature (from hipcc diagnostic): v4i* in AS1 (global) source,
// v4i* in AS3 (LDS) destination, imm offset, imm cpol.
// ---------------------------------------------------------------------------
#if defined(__has_builtin)
#  if __has_builtin(__builtin_amdgcn_global_load_async_to_lds_b128) && \
      __has_builtin(__builtin_amdgcn_s_wait_asynccnt)
#    define HAVE_ASYNC 1
#  endif
#endif
#ifndef HAVE_ASYNC
#  define HAVE_ASYNC 0
#endif

#if HAVE_ASYNC
__device__ __forceinline__ void async_copy16(const f16* g, f16* l) {
  __builtin_amdgcn_global_load_async_to_lds_b128(
      (__attribute__((address_space(1))) v4i*)(f16*)g,
      (__attribute__((address_space(3))) v4i*)l, 0, 0);
}
__device__ __forceinline__ void wait_async() {
  __builtin_amdgcn_s_wait_asynccnt(0);
}
#else
__device__ __forceinline__ void async_copy16(const f16* g, f16* l) {
  *(v8h*)l = *(const v8h*)g;   // synchronous fallback (global load + ds store)
}
__device__ __forceinline__ void wait_async() {}
#endif

// ---------------------------------------------------------------------------
// WMMA fragment helpers (layouts per CDNA5 ISA §7.12.2, wave32)
// ---------------------------------------------------------------------------

// A fragment: 16x32 f16, from row-major [*, ld] at (row0, k0).
// lane L: row = row0 + (L&15); halves 0..7 -> k0 + 8*(L>>4) + i,
//         halves 8..15 -> k0 + 16 + 8*(L>>4) + (i-8).
__device__ __forceinline__ v16h load_a_frag(const f16* __restrict__ A, int ld,
                                            int row0, int k0) {
  const int lane = threadIdx.x & 31;
  const int r = lane & 15;
  const int h = lane >> 4;
  const f16* p = A + (long)(row0 + r) * ld + k0 + 8 * h;
  v8h lo = *(const v8h*)(p);
  v8h hi = *(const v8h*)(p + 16);
  v16h a;
#pragma unroll
  for (int i = 0; i < 8; ++i) { a[i] = lo[i]; a[i + 8] = hi[i]; }
  return a;
}

// B fragment: 32x16 f16 (K x N) taken from "Bt" stored row-major [N, ld]:
// element(k,n) = Bt[n][k].  lane L: col = n0 + (L&15);
// halves 0..7 -> k0 + 8*(L>>4) + i, halves 8..15 -> k0 + 16 + 8*(L>>4) + (i-8).
__device__ __forceinline__ v16h load_b_frag(const f16* __restrict__ Bt, int ld,
                                            int n0, int k0) {
  const int lane = threadIdx.x & 31;
  const int n = lane & 15;
  const int h = lane >> 4;
  const f16* p = Bt + (long)(n0 + n) * ld + k0 + 8 * h;
  v8h lo = *(const v8h*)(p);
  v8h hi = *(const v8h*)(p + 16);
  v16h b;
#pragma unroll
  for (int i = 0; i < 8; ++i) { b[i] = lo[i]; b[i + 8] = hi[i]; }
  return b;
}

__device__ __forceinline__ v8f wmma_f16(v16h a, v16h b, v8f c) {
  // D = A x B + C ; emits v_wmma_f32_16x16x32_f16
  return __builtin_amdgcn_wmma_f32_16x16x32_f16(
      /*neg_a=*/false, a, /*neg_b=*/false, b,
      /*c_mod=*/(short)0, c, /*reuse_a=*/false, /*reuse_b=*/false);
}

// ---------------------------------------------------------------------------
// Kernel 0a: fp32 -> f16 weight conversion (Wq|Wk|Wv packed, Wp separate)
// ---------------------------------------------------------------------------
__global__ void wconv_kernel(const float* __restrict__ Wq,
                             const float* __restrict__ Wk,
                             const float* __restrict__ Wv,
                             const float* __restrict__ Wp,
                             f16* __restrict__ Wqkv, f16* __restrict__ Wp16) {
  long i = (long)blockIdx.x * blockDim.x + threadIdx.x;
  if (i < (long)Cn * Cn) {
    Wqkv[i]                      = (f16)Wq[i];
    Wqkv[(long)Cn * Cn + i]      = (f16)Wk[i];
    Wqkv[2l * Cn * Cn + i]       = (f16)Wv[i];
    Wp16[i]                      = (f16)Wp[i];
  }
}

// ---------------------------------------------------------------------------
// Kernel 0b: x [B,C,T] fp32 -> xT [B,T,C] f16 (LDS-tiled transpose)
// ---------------------------------------------------------------------------
__global__ __launch_bounds__(256) void xpose_kernel(const float* __restrict__ x,
                                                    f16* __restrict__ xT) {
  __shared__ float tile[32][33];
  const int b  = blockIdx.z;
  const int c0 = blockIdx.y * 32;
  const int t0 = blockIdx.x * 32;
  const int tx = threadIdx.x & 31;
  const int ty = threadIdx.x >> 5;  // 0..7
  const float* xb = x + (long)b * Cn * Tn;
#pragma unroll
  for (int i = 0; i < 32; i += 8)
    tile[ty + i][tx] = xb[(long)(c0 + ty + i) * Tn + t0 + tx];
  __syncthreads();
  f16* xtb = xT + (long)b * Tn * Cn;
#pragma unroll
  for (int i = 0; i < 32; i += 8)
    xtb[(long)(t0 + ty + i) * Cn + c0 + tx] = (f16)tile[tx][ty + i];
}

// ---------------------------------------------------------------------------
// Kernel 1: fused QKV projection.  out[o,t] = sum_c W[o,c] * x[b][c,t]
// A = W (row-major [C,C] f16), Bt = xT[b] ([T,C] f16).
// Block: 256 thr = 8 waves (4 M x 2 N); wave tile 32x64 (2 A-frags, 4 B-frags,
// 8 WMMAs per k-step); block tile = 128 x 128.  grid.z = 3*B.
// Q,K stored [B,H,T,D]; V stored transposed [B,H,D,T].
// ---------------------------------------------------------------------------
__device__ __forceinline__ void store_head_tile(f16* __restrict__ dst, int b,
                                                int o_base, int t_base,
                                                const v8f& c, bool vmode) {
  const int lane = threadIdx.x & 31;
  const int t  = t_base + (lane & 15);
  const int hh = lane >> 4;
#pragma unroll
  for (int i = 0; i < 8; ++i) {
    const int o = o_base + i + 8 * hh;
    const int head = o >> 6;      // o / D
    const int d    = o & 63;      // o % D
    long idx = vmode ? ((((long)b * Hn + head) * Dn + d) * Tn + t)
                     : ((((long)b * Hn + head) * Tn + t) * Dn + d);
    dst[idx] = (f16)c[i];
  }
}

__global__ __launch_bounds__(256) void qkv_gemm_kernel(
    const f16* __restrict__ Wqkv, const f16* __restrict__ xT,
    f16* __restrict__ Q, f16* __restrict__ K, f16* __restrict__ Vt) {
  const int wid = threadIdx.x >> 5;
  const int mw = wid & 3;           // 4 waves along M
  const int nw = wid >> 2;          // 2 waves along N
  const int z = blockIdx.z;
  const int which = z % 3;
  const int b = z / 3;
  const int m0 = blockIdx.y * 128 + mw * 32;
  const int n0 = blockIdx.x * 128 + nw * 64;
  const f16* A  = Wqkv + (long)which * Cn * Cn;
  const f16* Bt = xT + (long)b * Tn * Cn;

  const v8f vzero = {};
  v8f c[2][4];
#pragma unroll
  for (int i = 0; i < 2; ++i)
#pragma unroll
    for (int j = 0; j < 4; ++j) c[i][j] = vzero;

  for (int k0 = 0; k0 < Cn; k0 += 32) {
    if (k0 + 32 < Cn)   // near-scope (WGP) prefetch of next k-tile: operands
      __builtin_prefetch(Bt + (long)n0 * Cn + k0 + 32, 0, 3);  // live in L2
    v16h a0 = load_a_frag(A, Cn, m0, k0);
    v16h a1 = load_a_frag(A, Cn, m0 + 16, k0);
#pragma unroll
    for (int j = 0; j < 4; ++j) {
      v16h bf = load_b_frag(Bt, Cn, n0 + 16 * j, k0);
      c[0][j] = wmma_f16(a0, bf, c[0][j]);
      c[1][j] = wmma_f16(a1, bf, c[1][j]);
    }
  }
  f16* dst = (which == 0) ? Q : (which == 1) ? K : Vt;
  const bool vmode = (which == 2);
#pragma unroll
  for (int i = 0; i < 2; ++i)
#pragma unroll
    for (int j = 0; j < 4; ++j)
      store_head_tile(dst, b, m0 + 16 * i, n0 + 16 * j, c[i][j], vmode);
}

// ---------------------------------------------------------------------------
// Kernel 2: flash attention per (b,h).  One wave owns 16 query rows; 8 waves
// per block -> 128 rows/block.  The 32-key K tile (32x64) and V tile (64x32)
// are shared by all 8 waves, so they are staged once per block into LDS with
// the CDNA5 async global->LDS path, double-buffered against compute.
// Per 32-key step: 4 WMMAs for energy, online softmax with shfl_xor row
// reductions, P bounced through LDS (C-layout -> A-layout), 4 WMMAs for P*V.
// Output -> merged [B,T,C] f16 (head h occupies cols h*D .. h*D+63).
// ---------------------------------------------------------------------------
__global__ __launch_bounds__(256) void flash_attn_kernel(
    const f16* __restrict__ Q, const f16* __restrict__ K,
    const f16* __restrict__ Vt, f16* __restrict__ merged) {
  __shared__ f16 Ktile[2][32 * 64];   // [s][d], ld=64, 4 KB per buffer
  __shared__ f16 Vtile[2][64 * 32];   // [d][s], ld=32, 4 KB per buffer
  __shared__ f16 plds[8][16 * 32];    // per-wave P staging (1 KB each)

  const int tid  = threadIdx.x;
  const int wid  = tid >> 5;
  const int lane = tid & 31;
  const int ncol = lane & 15;
  const int hh   = lane >> 4;
  const int bh = blockIdx.y;           // b*H + h
  const int b  = bh / Hn;
  const int hd = bh % Hn;
  const int t0 = blockIdx.x * 128 + wid * 16;

  const f16* Qbh = Q  + (long)bh * Tn * Dn;   // [T,D]
  const f16* Kbh = K  + (long)bh * Tn * Dn;   // [T,D]
  const f16* Vbh = Vt + (long)bh * Dn * Tn;   // [D,T]

  // cooperative tile staging: 256 threads, one 16B chunk each per tile
  const int krow = tid >> 3, kcol = (tid & 7) * 8;   // K: 32 rows x 128B
  const int vrow = tid >> 2, vcol = (tid & 3) * 8;   // V: 64 rows x 64B
  auto stage = [&](int buf, int s0) {
    async_copy16(Kbh + (long)(s0 + krow) * Dn + kcol,
                 &Ktile[buf][krow * 64 + kcol]);
    async_copy16(Vbh + (long)vrow * Tn + s0 + vcol,
                 &Vtile[buf][vrow * 32 + vcol]);
  };

  // Q fragments held for the whole pass (d = 0..31, 32..63)
  const v16h qa0 = load_a_frag(Qbh, Dn, t0, 0);
  const v16h qa1 = load_a_frag(Qbh, Dn, t0, 32);

  v8f acc0 = {}, acc1 = {}, acc2 = {}, acc3 = {};   // out tile 16 x 64
  float m[8], l[8];
#pragma unroll
  for (int j = 0; j < 8; ++j) { m[j] = -1e30f; l[j] = 0.0f; }
  const float inv_scale = 0.125f;     // 1/sqrt(D)
  f16* Pw = plds[wid];

  stage(0, 0);                        // prologue: tile 0 in flight
  for (int s0 = 0; s0 < Tn; s0 += 32) {
    const int buf = (s0 >> 5) & 1;
    wait_async();                     // my chunks of tile s0 have landed
    __syncthreads();                  // everyone's chunks landed; prev compute done
    if (s0 + 32 < Tn) stage(buf ^ 1, s0 + 32);  // overlap next tile with compute

    // K fragments from LDS tile (ds loads; shared across all 8 waves)
    v16h kb00 = load_b_frag(Ktile[buf], 64, 0,  0);
    v16h kb01 = load_b_frag(Ktile[buf], 64, 0,  32);
    v16h kb10 = load_b_frag(Ktile[buf], 64, 16, 0);
    v16h kb11 = load_b_frag(Ktile[buf], 64, 16, 32);
    v8f e0 = {}, e1 = {};
    e0 = wmma_f16(qa0, kb00, e0);
    e0 = wmma_f16(qa1, kb01, e0);
    e1 = wmma_f16(qa0, kb10, e1);
    e1 = wmma_f16(qa1, kb11, e1);

    // online softmax; row = j + 8*hh, cols live across the 16-lane group
#pragma unroll
    for (int j = 0; j < 8; ++j) {
      float x0 = e0[j] * inv_scale;
      float x1 = e1[j] * inv_scale;
      float rmax = fmaxf(x0, x1);
      rmax = fmaxf(rmax, __shfl_xor(rmax, 1));
      rmax = fmaxf(rmax, __shfl_xor(rmax, 2));
      rmax = fmaxf(rmax, __shfl_xor(rmax, 4));
      rmax = fmaxf(rmax, __shfl_xor(rmax, 8));
      const float mn = fmaxf(m[j], rmax);
      const float sc = __expf(m[j] - mn);
      m[j] = mn;
      const float p0 = __expf(x0 - mn);
      const float p1 = __expf(x1 - mn);
      float rs = p0 + p1;
      rs += __shfl_xor(rs, 1);
      rs += __shfl_xor(rs, 2);
      rs += __shfl_xor(rs, 4);
      rs += __shfl_xor(rs, 8);
      l[j] = l[j] * sc + rs;
      acc0[j] *= sc; acc1[j] *= sc; acc2[j] *= sc; acc3[j] *= sc;
      const int row = j + 8 * hh;
      Pw[row * 32 + ncol]      = (f16)p0;
      Pw[row * 32 + 16 + ncol] = (f16)p1;
    }
    // P (16x32) back as A fragment from LDS (wave-private region, ds in-order)
    v16h pa = load_a_frag(Pw, 32, 0, 0);
    // V fragments from LDS tile: out cols d0..d0+15, k = local s 0..31
    v16h vb0 = load_b_frag(Vtile[buf], 32, 0,  0);
    v16h vb1 = load_b_frag(Vtile[buf], 32, 16, 0);
    v16h vb2 = load_b_frag(Vtile[buf], 32, 32, 0);
    v16h vb3 = load_b_frag(Vtile[buf], 32, 48, 0);
    acc0 = wmma_f16(pa, vb0, acc0);
    acc1 = wmma_f16(pa, vb1, acc1);
    acc2 = wmma_f16(pa, vb2, acc2);
    acc3 = wmma_f16(pa, vb3, acc3);
  }

  // normalize and store merged[b, t, hd*D + d]
  f16* mrow = merged + (long)b * Tn * Cn + (long)hd * Dn;
#pragma unroll
  for (int j = 0; j < 8; ++j) {
    const int t = t0 + j + 8 * hh;
    const float invl = 1.0f / l[j];
    f16* p = mrow + (long)t * Cn;
    p[ncol]      = (f16)(acc0[j] * invl);
    p[16 + ncol] = (f16)(acc1[j] * invl);
    p[32 + ncol] = (f16)(acc2[j] * invl);
    p[48 + ncol] = (f16)(acc3[j] * invl);
  }
}

// ---------------------------------------------------------------------------
// Kernel 3: output projection.  out[b,o,t] = sum_c Wp[o,c] * merged[b,t,c]
// A = Wp16 [C,C], Bt = merged[b] [T,C].  fp32 output [B,C,T].
// Same 32x64 wave tile / 128x128 block tile as kernel 1.
// ---------------------------------------------------------------------------
__device__ __forceinline__ void store_out_tile(float* __restrict__ out, int b,
                                               int o_base, int t_base,
                                               const v8f& c) {
  const int lane = threadIdx.x & 31;
  const int t  = t_base + (lane & 15);
  const int hh = lane >> 4;
#pragma unroll
  for (int i = 0; i < 8; ++i) {
    const int o = o_base + i + 8 * hh;
    out[((long)b * Cn + o) * Tn + t] = c[i];
  }
}

__global__ __launch_bounds__(256) void proj_gemm_kernel(
    const f16* __restrict__ Wp16, const f16* __restrict__ merged,
    float* __restrict__ out) {
  const int wid = threadIdx.x >> 5;
  const int mw = wid & 3;
  const int nw = wid >> 2;
  const int b = blockIdx.z;
  const int m0 = blockIdx.y * 128 + mw * 32;
  const int n0 = blockIdx.x * 128 + nw * 64;
  const f16* Bt = merged + (long)b * Tn * Cn;

  const v8f vzero = {};
  v8f c[2][4];
#pragma unroll
  for (int i = 0; i < 2; ++i)
#pragma unroll
    for (int j = 0; j < 4; ++j) c[i][j] = vzero;

  for (int k0 = 0; k0 < Cn; k0 += 32) {
    if (k0 + 32 < Cn)
      __builtin_prefetch(Bt + (long)n0 * Cn + k0 + 32, 0, 3);
    v16h a0 = load_a_frag(Wp16, Cn, m0, k0);
    v16h a1 = load_a_frag(Wp16, Cn, m0 + 16, k0);
#pragma unroll
    for (int j = 0; j < 4; ++j) {
      v16h bf = load_b_frag(Bt, Cn, n0 + 16 * j, k0);
      c[0][j] = wmma_f16(a0, bf, c[0][j]);
      c[1][j] = wmma_f16(a1, bf, c[1][j]);
    }
  }
#pragma unroll
  for (int i = 0; i < 2; ++i)
#pragma unroll
    for (int j = 0; j < 4; ++j)
      store_out_tile(out, b, m0 + 16 * i, n0 + 16 * j, c[i][j]);
}

// ---------------------------------------------------------------------------
// Launch
// ---------------------------------------------------------------------------
extern "C" void kernel_launch(void* const* d_in, const int* in_sizes, int n_in,
                              void* d_out, int out_size, void* d_ws,
                              size_t ws_size, hipStream_t stream) {
  const float* x  = (const float*)d_in[0];
  const float* Wq = (const float*)d_in[1];
  const float* Wk = (const float*)d_in[2];
  const float* Wv = (const float*)d_in[3];
  const float* Wp = (const float*)d_in[4];
  float* out = (float*)d_out;

  // workspace layout (all f16, offsets in bytes, 256B-aligned)
  char* ws = (char*)d_ws;
  const long CC   = (long)Cn * Cn;           // 1M elems
  const long BTC  = (long)Bn * Tn * Cn;      // 8M elems
  f16* Wqkv   = (f16*)(ws);                               // 3*CC = 6 MB
  f16* Wp16   = (f16*)(ws + 3 * CC * 2);                  // CC   = 2 MB
  f16* xT     = (f16*)(ws + 4 * CC * 2);                  // BTC  = 16 MB
  f16* Qh     = (f16*)(ws + 4 * CC * 2 + 1 * BTC * 2);    // 16 MB
  f16* Kh     = (f16*)(ws + 4 * CC * 2 + 2 * BTC * 2);    // 16 MB
  f16* Vt     = (f16*)(ws + 4 * CC * 2 + 3 * BTC * 2);    // 16 MB
  f16* merged = (f16*)(ws + 4 * CC * 2 + 4 * BTC * 2);    // 16 MB

  // 0) precision conversion + transpose
  wconv_kernel<<<dim3((CC + 255) / 256), 256, 0, stream>>>(Wq, Wk, Wv, Wp,
                                                           Wqkv, Wp16);
  xpose_kernel<<<dim3(Tn / 32, Cn / 32, Bn), 256, 0, stream>>>(x, xT);

  // 1) fused QKV projection
  qkv_gemm_kernel<<<dim3(Tn / 128, Cn / 128, 3 * Bn), 256, 0, stream>>>(
      Wqkv, xT, Qh, Kh, Vt);

  // 2) flash attention (async LDS staging of K/V tiles)
  flash_attn_kernel<<<dim3(Tn / 128, Bn * Hn), 256, 0, stream>>>(Qh, Kh, Vt,
                                                                 merged);

  // 3) output projection -> fp32 [B,C,T]
  proj_gemm_kernel<<<dim3(Tn / 128, Cn / 128, Bn), 256, 0, stream>>>(Wp16,
                                                                     merged,
                                                                     out);
}
